// TransformerBlock_71012989272467
// MI455X (gfx1250) — matrix operand order repaired and checked by reading, not compiled, not run on hardware
//
#include <hip/hip_runtime.h>
#include <hip/hip_bf16.h>
#include <math.h>
#include <stdint.h>

// ---------------- problem constants ----------------
#define B_    8
#define N_    1024
#define C_    768
#define H_    12
#define D_    64
#define HID_  3072
#define MTOT  (B_ * N_)        // 8192 rows
#define RS_   (3 * C_)         // qkv row stride = 2304

typedef __attribute__((ext_vector_type(16))) __bf16 v16bf;
typedef __attribute__((ext_vector_type(8)))  float  v8f;
typedef int v4i_ __attribute__((__vector_size__(16)));

union F8   { v8f   v; float f[8]; };
union AB32 { uint4 u[2]; v16bf v; };   // 32 bytes = one 16x32 bf16 fragment

// ---- gfx1250 async global->LDS path (guarded; fallback = load+store) ----
#if defined(__HIP_DEVICE_COMPILE__) && \
    __has_builtin(__builtin_amdgcn_global_load_async_to_lds_b128) && \
    __has_builtin(__builtin_amdgcn_s_wait_asynccnt)
#define HAS_ASYNC 1
#else
#define HAS_ASYNC 0
#endif

#if HAS_ASYNC
typedef v4i_ __attribute__((address_space(3)))* lds_v4i_ptr;
typedef v4i_ __attribute__((address_space(1)))* glb_v4i_ptr;
__device__ __forceinline__ lds_v4i_ptr to_lds(void* p) {
  // generic LDS address truncates to 32-bit LDS offset
  return (lds_v4i_ptr)(uint32_t)(uintptr_t)p;
}
__device__ __forceinline__ glb_v4i_ptr to_glb(const void* p) {
  return (glb_v4i_ptr)(uintptr_t)p;
}
#define ASYNC_COPY16(gsrc, ldst) \
  __builtin_amdgcn_global_load_async_to_lds_b128(to_glb(gsrc), to_lds(ldst), 0, 0)
#define ASYNC_WAIT0() __builtin_amdgcn_s_wait_asynccnt(0)
#else
#define ASYNC_COPY16(gsrc, ldst) (*(uint4*)(ldst) = *(const uint4*)(gsrc))
#define ASYNC_WAIT0() ((void)0)
#endif

__device__ __forceinline__ unsigned short f2bf(float f) {
  unsigned int u = __float_as_uint(f);
  u += 0x7FFFu + ((u >> 16) & 1u);     // round-to-nearest-even
  return (unsigned short)(u >> 16);
}

__device__ __forceinline__ v8f wmma_bf16(const AB32& a, const AB32& b, v8f c) {
  // (neg_a, A, neg_b, B, c_mod, C, reuse_a, reuse_b)
  return __builtin_amdgcn_wmma_f32_16x16x32_bf16(false, a.v, false, b.v,
                                                 (short)0, c, false, false);
}

// ---------------- fp32 -> bf16 convert ----------------
__global__ __launch_bounds__(256)
void cvt_kernel(const float* __restrict__ in, unsigned short* __restrict__ out, int n) {
  int i = blockIdx.x * 256 + threadIdx.x;
  if (i < n) out[i] = f2bf(in[i]);
}

// ---------------- LayerNorm -> bf16 ----------------
__global__ __launch_bounds__(256)
void ln_kernel(const float* __restrict__ x, const float* __restrict__ w,
               const float* __restrict__ bgain, unsigned short* __restrict__ out) {
  __shared__ float s1[256];
  __shared__ float s2[256];
  const int row = blockIdx.x;
  const int t = threadIdx.x;
  const float* xr = x + (size_t)row * C_;
  float v0 = xr[t], v1 = xr[t + 256], v2 = xr[t + 512];
  s1[t] = v0 + v1 + v2;
  s2[t] = v0 * v0 + v1 * v1 + v2 * v2;
  __syncthreads();
  for (int off = 128; off > 0; off >>= 1) {
    if (t < off) { s1[t] += s1[t + off]; s2[t] += s2[t + off]; }
    __syncthreads();
  }
  float mu  = s1[0] * (1.0f / C_);
  float var = s2[0] * (1.0f / C_) - mu * mu;
  float rs  = rsqrtf(var + 1e-5f);
  unsigned short* orow = out + (size_t)row * C_;
  orow[t]       = f2bf((v0 - mu) * rs * w[t]       + bgain[t]);
  orow[t + 256] = f2bf((v1 - mu) * rs * w[t + 256] + bgain[t + 256]);
  orow[t + 512] = f2bf((v2 - mu) * rs * w[t + 512] + bgain[t + 512]);
}

// ---------------- GEMM: D[M,O] = A[M,K] @ W[O,K]^T + bias ----------------
// Workgroup tile 128x128, Ktile 64, 8 waves in 4x2 grid, 32x64 per wave.
// Double-buffered LDS; global->LDS staging uses GLOBAL_LOAD_ASYNC_TO_LDS_B128.
// MODE 0: bf16 out   MODE 1: GELU -> bf16 out   MODE 2: fp32 out + residual
template <int MODE>
__global__ __launch_bounds__(256)
void gemm_kernel(const unsigned short* __restrict__ A,
                 const unsigned short* __restrict__ W,
                 const float* __restrict__ bias,
                 const float* __restrict__ residual,
                 unsigned short* __restrict__ outB,
                 float* __restrict__ outF,
                 int K, int O) {
  __shared__ __align__(16) unsigned short As[2][128 * 72];   // 128x64, stride 72
  __shared__ __align__(16) unsigned short Bs[2][128 * 72];
  const int t     = threadIdx.x;
  const int wave  = t >> 5, lane = t & 31;
  const int lhalf = lane >> 4, l16 = lane & 15;
  const int wm = wave >> 1, wn = wave & 1;     // 4x2 wave grid
  const int blockM = blockIdx.y * 128;
  const int blockN = blockIdx.x * 128;

  F8 acc[2][4];
#pragma unroll
  for (int i = 0; i < 2; i++)
#pragma unroll
    for (int j = 0; j < 4; j++)
#pragma unroll
      for (int r = 0; r < 8; r++) acc[i][j].f[r] = 0.0f;

  // stage one 128x64 K-slab of A and W into LDS buffer `buf`
  auto stage = [&](int k0, int buf) {
#pragma unroll
    for (int i = 0; i < 4; i++) {
      int c = t + i * 256;                 // 1024 chunks of 8 halves
      int row = c >> 3, k8 = c & 7;
      const unsigned short* ga = A + (size_t)(blockM + row) * K + k0 + k8 * 8;
      const unsigned short* gw = W + (size_t)(blockN + row) * K + k0 + k8 * 8;
      ASYNC_COPY16(ga, &As[buf][row * 72 + k8 * 8]);
      ASYNC_COPY16(gw, &Bs[buf][row * 72 + k8 * 8]);
    }
  };

  stage(0, 0);
  int cur = 0;
  for (int k0 = 0; k0 < K; k0 += 64) {
    ASYNC_WAIT0();          // this wave's copies into As/Bs[cur] are done
    __syncthreads();        // everyone's copies visible
    if (k0 + 64 < K) {
      stage(k0 + 64, cur ^ 1);                               // overlap with compute
      __builtin_prefetch(A + (size_t)(blockM + (t >> 1)) * K + k0 + 128, 0, 0);
    }
#pragma unroll
    for (int ks = 0; ks < 2; ks++) {
      AB32 af[2], bf[4];
#pragma unroll
      for (int mi = 0; mi < 2; mi++) {
        // A 16x32 bf16 layout: lane<16 holds K {0..7,16..23}, lane>=16 {8..15,24..31}
        const unsigned short* p =
            &As[cur][(wm * 32 + mi * 16 + l16) * 72 + ks * 32 + lhalf * 8];
        af[mi].u[0] = *(const uint4*)(p);
        af[mi].u[1] = *(const uint4*)(p + 16);
      }
#pragma unroll
      for (int ni = 0; ni < 4; ni++) {
        // B 32x16 bf16 layout: lane<16 holds K {0..7,16..23} of col N=l16, lane>=16 {8..15,24..31}
        const unsigned short* p =
            &Bs[cur][(wn * 64 + ni * 16 + l16) * 72 + ks * 32 + lhalf * 8];
        bf[ni].u[0] = *(const uint4*)(p);
        bf[ni].u[1] = *(const uint4*)(p + 16);
      }
#pragma unroll
      for (int mi = 0; mi < 2; mi++)
#pragma unroll
        for (int ni = 0; ni < 4; ni++)
          acc[mi][ni].v = wmma_bf16(af[mi], bf[ni], acc[mi][ni].v);
    }
    __syncthreads();        // readers done before buffer reuse
    cur ^= 1;
  }

  // epilogue: C/D layout -> lane holds col N=l16, VGPR r holds row r + 8*lhalf
#pragma unroll
  for (int mi = 0; mi < 2; mi++) {
#pragma unroll
    for (int ni = 0; ni < 4; ni++) {
      int col = blockN + wn * 64 + ni * 16 + l16;
      float bv = bias[col];
#pragma unroll
      for (int r = 0; r < 8; r++) {
        int row = blockM + wm * 32 + mi * 16 + r + 8 * lhalf;
        float v = acc[mi][ni].f[r] + bv;
        size_t idx = (size_t)row * O + col;
        if (MODE == 2) {
          outF[idx] = residual[idx] + v;
        } else if (MODE == 1) {
          outB[idx] = f2bf(0.5f * v * (1.0f + erff(v * 0.70710678118f)));
        } else {
          outB[idx] = f2bf(v);
        }
      }
    }
  }
}

// ---------------- flash attention ----------------
// grid: (N/64, B*H), block: 128 (4 waves). Each wave owns 16 query rows.
__global__ __launch_bounds__(128)
void attn_kernel(const unsigned short* __restrict__ qkv,
                 unsigned short* __restrict__ attn_out) {
  __shared__ __align__(16) unsigned short Ks[32 * 72];      // K tile, row-major
  __shared__ __align__(16) unsigned short Vt[64 * 40];      // V^T: [d][key], stride 40
  __shared__ __align__(16) unsigned short Pl[4][16 * 40];   // per-wave P tile
  const int bh = blockIdx.y;
  const int b = bh / H_, h = bh % H_;
  const int rowTile = blockIdx.x;
  const int t = threadIdx.x;
  const int wave = t >> 5, lane = t & 31;
  const int lhalf = lane >> 4, l16 = lane & 15;
  const size_t seq0 = (size_t)b * N_;

  const unsigned short* Qb = qkv + (seq0 + rowTile * 64 + wave * 16 + l16) * RS_ + h * D_;
  const unsigned short* Kb = qkv + seq0 * RS_ + (H_ + h) * D_;
  const unsigned short* Vb = qkv + seq0 * RS_ + (2 * H_ + h) * D_;

  // Q fragments (A-matrix, contraction over d=64 in two x32 steps), loaded once
  AB32 qf[2];
#pragma unroll
  for (int ks = 0; ks < 2; ks++) {
    const unsigned short* p = Qb + ks * 32 + lhalf * 8;
    qf[ks].u[0] = *(const uint4*)(p);
    qf[ks].u[1] = *(const uint4*)(p + 16);
  }

  F8 o[4];
  float rmax[8], rsum[8];
#pragma unroll
  for (int d = 0; d < 4; d++)
#pragma unroll
    for (int r = 0; r < 8; r++) o[d].f[r] = 0.0f;
#pragma unroll
  for (int r = 0; r < 8; r++) { rmax[r] = -1e30f; rsum[r] = 0.0f; }

  unsigned short* P = Pl[wave];

  for (int kt = 0; kt < 32; kt++) {
    __syncthreads();                       // previous iteration's readers done
    // stage K tile (shared across waves) via async global->LDS copies
#pragma unroll
    for (int i = 0; i < 2; i++) {
      int c = t + i * 128;                 // 256 chunks of 8 halves
      int row = c >> 3, k8 = c & 7;
      ASYNC_COPY16(Kb + (size_t)(kt * 32 + row) * RS_ + k8 * 8,
                   &Ks[row * 72 + k8 * 8]);
    }
    // stage V tile transposed: Vt[d][key] (manual; transpose in the store)
#pragma unroll
    for (int i = 0; i < 2; i++) {
      int c = t + i * 128;
      int j = c >> 3, d8 = c & 7;
      uint4 vv = *(const uint4*)(Vb + (size_t)(kt * 32 + j) * RS_ + d8 * 8);
      const unsigned short* pv = (const unsigned short*)&vv;
#pragma unroll
      for (int e = 0; e < 8; e++) Vt[(d8 * 8 + e) * 40 + j] = pv[e];
    }
    ASYNC_WAIT0();
    __syncthreads();

    // S = Q K^T : two 16x16 D tiles (32 keys)
    F8 s[2];
#pragma unroll
    for (int ns = 0; ns < 2; ns++) {
#pragma unroll
      for (int r = 0; r < 8; r++) s[ns].f[r] = 0.0f;
#pragma unroll
      for (int ks = 0; ks < 2; ks++) {
        AB32 kf;
        const unsigned short* p = Ks + (ns * 16 + l16) * 72 + ks * 32 + lhalf * 8;
        kf.u[0] = *(const uint4*)(p);
        kf.u[1] = *(const uint4*)(p + 16);
        s[ns].v = wmma_bf16(qf[ks], kf, s[ns].v);
      }
    }

    // online softmax; rows spread per-VGPR, cols across 16-lane groups
#pragma unroll
    for (int r = 0; r < 8; r++) {
      float s0 = s[0].f[r] * 0.125f;       // scale = D^-0.5
      float s1 = s[1].f[r] * 0.125f;
      float mx = fmaxf(s0, s1);
#pragma unroll
      for (int off = 8; off >= 1; off >>= 1) mx = fmaxf(mx, __shfl_xor(mx, off, 16));
      float nm = fmaxf(rmax[r], mx);
      float corr = __expf(rmax[r] - nm);
      rmax[r] = nm;
      float p0 = __expf(s0 - nm), p1 = __expf(s1 - nm);
      float ts = p0 + p1;
#pragma unroll
      for (int off = 8; off >= 1; off >>= 1) ts += __shfl_xor(ts, off, 16);
      rsum[r] = rsum[r] * corr + ts;
#pragma unroll
      for (int d = 0; d < 4; d++) o[d].f[r] *= corr;
      int prow = (r + 8 * lhalf) * 40;
      P[prow + l16]      = f2bf(p0);
      P[prow + 16 + l16] = f2bf(p1);
    }

    // O += P @ V  (wave-private P; same-wave LDS ops are in order)
    AB32 pf;
    pf.u[0] = *(const uint4*)(P + l16 * 40 + lhalf * 8);
    pf.u[1] = *(const uint4*)(P + l16 * 40 + 16 + lhalf * 8);
#pragma unroll
    for (int d = 0; d < 4; d++) {
      AB32 vf;
      const unsigned short* p = Vt + (d * 16 + l16) * 40 + lhalf * 8;
      vf.u[0] = *(const uint4*)(p);
      vf.u[1] = *(const uint4*)(p + 16);
      o[d].v = wmma_bf16(pf, vf, o[d].v);
    }
  }

  // normalize and store bf16 at [b,n, h*64+d]
#pragma unroll
  for (int r = 0; r < 8; r++) {
    float inv = 1.0f / rsum[r];
    int row = rowTile * 64 + wave * 16 + r + 8 * lhalf;
#pragma unroll
    for (int d = 0; d < 4; d++) {
      attn_out[(seq0 + row) * C_ + h * D_ + d * 16 + l16] = f2bf(o[d].f[r] * inv);
    }
  }
}

// ---------------- launcher ----------------
extern "C" void kernel_launch(void* const* d_in, const int* in_sizes, int n_in,
                              void* d_out, int out_size, void* d_ws, size_t ws_size,
                              hipStream_t stream) {
  (void)in_sizes; (void)n_in; (void)out_size; (void)ws_size;
  const float* x      = (const float*)d_in[0];
  const float* ln1_w  = (const float*)d_in[1];
  const float* ln1_b  = (const float*)d_in[2];
  const float* qkv_w  = (const float*)d_in[3];
  const float* qkv_b  = (const float*)d_in[4];
  const float* proj_w = (const float*)d_in[5];
  const float* proj_b = (const float*)d_in[6];
  const float* ln2_w  = (const float*)d_in[7];
  const float* ln2_b  = (const float*)d_in[8];
  const float* fc1_w  = (const float*)d_in[9];
  const float* fc1_b  = (const float*)d_in[10];
  const float* fc2_w  = (const float*)d_in[11];
  const float* fc2_b  = (const float*)d_in[12];

  char* ws = (char*)d_ws;
  auto alloc = [&](size_t bytes) -> char* {
    char* p = ws;
    ws += (bytes + 255) & ~(size_t)255;
    return p;
  };
  unsigned short* wq   = (unsigned short*)alloc((size_t)RS_ * C_ * 2);       // 2304x768
  unsigned short* wp   = (unsigned short*)alloc((size_t)C_ * C_ * 2);        // 768x768
  unsigned short* w1   = (unsigned short*)alloc((size_t)HID_ * C_ * 2);      // 3072x768
  unsigned short* w2   = (unsigned short*)alloc((size_t)C_ * HID_ * 2);      // 768x3072
  unsigned short* h1   = (unsigned short*)alloc((size_t)MTOT * C_ * 2);
  unsigned short* qkvb = (unsigned short*)alloc((size_t)MTOT * RS_ * 2);
  unsigned short* atto = (unsigned short*)alloc((size_t)MTOT * C_ * 2);
  float*          x1   = (float*)alloc((size_t)MTOT * C_ * 4);
  unsigned short* h2   = (unsigned short*)alloc((size_t)MTOT * C_ * 2);
  unsigned short* f1o  = (unsigned short*)alloc((size_t)MTOT * HID_ * 2);

  // bf16 weight copies
  {
    int n;
    n = RS_ * C_;   cvt_kernel<<<(n + 255) / 256, 256, 0, stream>>>(qkv_w,  wq, n);
    n = C_ * C_;    cvt_kernel<<<(n + 255) / 256, 256, 0, stream>>>(proj_w, wp, n);
    n = HID_ * C_;  cvt_kernel<<<(n + 255) / 256, 256, 0, stream>>>(fc1_w,  w1, n);
    n = C_ * HID_;  cvt_kernel<<<(n + 255) / 256, 256, 0, stream>>>(fc2_w,  w2, n);
  }

  // LN1 -> h1 (bf16)
  ln_kernel<<<MTOT, 256, 0, stream>>>(x, ln1_w, ln1_b, h1);

  // QKV: [8192,768] x [2304,768]^T -> bf16 [8192,2304]
  gemm_kernel<0><<<dim3(RS_ / 128, MTOT / 128), 256, 0, stream>>>(
      h1, wq, qkv_b, nullptr, qkvb, nullptr, C_, RS_);

  // attention -> bf16 [8192,768]
  attn_kernel<<<dim3(N_ / 64, B_ * H_), 128, 0, stream>>>(qkvb, atto);

  // proj + residual(x) -> fp32 x1
  gemm_kernel<2><<<dim3(C_ / 128, MTOT / 128), 256, 0, stream>>>(
      atto, wp, proj_b, x, nullptr, x1, C_, C_);

  // LN2 -> h2 (bf16)
  ln_kernel<<<MTOT, 256, 0, stream>>>(x1, ln2_w, ln2_b, h2);

  // FC1 + GELU -> bf16 [8192,3072]
  gemm_kernel<1><<<dim3(HID_ / 128, MTOT / 128), 256, 0, stream>>>(
      h2, w1, fc1_b, nullptr, f1o, nullptr, C_, HID_);

  // FC2 + residual(x1) -> fp32 d_out
  gemm_kernel<2><<<dim3(C_ / 128, MTOT / 128), 256, 0, stream>>>(
      f1o, w2, fc2_b, x1, nullptr, (float*)d_out, HID_, C_);
}
